// TransolverBlock_15436112462085
// MI455X (gfx1250) — matrix operand order repaired and checked by reading, not compiled, not run on hardware
//
#include <hip/hip_runtime.h>
#include <math.h>

// Problem sizes (fixed by the reference)
#define Bn 4
#define Nn 32768
#define Cn 256
#define Hn 8
#define Dn 32
#define Gn 64
#define HIDn 1024
#define BNn (Bn*Nn)

// ---------------- CDNA5 WMMA plumbing (wave32, 16x16x32 bf16) ----------------
typedef __attribute__((ext_vector_type(16))) __bf16 v16bf;
typedef __attribute__((ext_vector_type(8)))  float  f32x8;
typedef __attribute__((ext_vector_type(8)))  unsigned int u32x8;
typedef __attribute__((ext_vector_type(4)))  unsigned int u32x4;
typedef __attribute__((ext_vector_type(8)))  int i32x8;
typedef __attribute__((ext_vector_type(4)))  int i32x4;

__device__ __forceinline__ unsigned short f2bf(float f){
  unsigned int u = __float_as_uint(f);
  u += 0x7FFFu + ((u >> 16) & 1u);          // round-to-nearest-even
  return (unsigned short)(u >> 16);
}
__device__ __forceinline__ unsigned int pack2(unsigned short a, unsigned short b){
  return (unsigned int)a | ((unsigned int)b << 16);
}
__device__ __forceinline__ float gelu_f(float v){
  return 0.5f * v * (1.0f + erff(v * 0.70710678118654752f));
}
__device__ __forceinline__ f32x8 fzero8(){
  f32x8 c = {0.f,0.f,0.f,0.f,0.f,0.f,0.f,0.f};
  return c;
}
// A-operand (16x32 bf16): per ISA, lane half h holds K = k0+8h+(0..7) and K = k0+16+8h+(0..7)
__device__ __forceinline__ v16bf load_a16(const unsigned short* rowptr, int k0, int half){
  const uint4* p0 = (const uint4*)(rowptr + k0 + 8*half);
  const uint4* p1 = (const uint4*)(rowptr + k0 + 16 + 8*half);
  uint4 a = *p0; uint4 b = *p1;
  u32x8 u = {a.x,a.y,a.z,a.w,b.x,b.y,b.z,b.w};
  return __builtin_bit_cast(v16bf, u);
}
// B-operand (32x16 bf16): lane half h holds K = k0+8h+(0..7) and K = k0+16+8h+(0..7), same as A
__device__ __forceinline__ v16bf load_b16(const unsigned short* rowptr, int k0, int half){
  const uint4* p0 = (const uint4*)(rowptr + k0 + 8*half);
  const uint4* p1 = (const uint4*)(rowptr + k0 + 16 + 8*half);
  uint4 a = *p0; uint4 b = *p1;
  u32x8 u = {a.x,a.y,a.z,a.w,b.x,b.y,b.z,b.w};
  return __builtin_bit_cast(v16bf, u);
}
__device__ __forceinline__ f32x8 wmma_bf(v16bf a, v16bf b, f32x8 c){
  return __builtin_amdgcn_wmma_f32_16x16x32_bf16(false, a, false, b, (short)0, c, false, false);
}

// ---------------- workspace layout (bytes) ----------------
#define OFF_WX     ((size_t)0)          // Wx bf16        131072
#define OFF_WFX    ((size_t)131072)     // Wfx bf16       131072
#define OFF_WOUT   ((size_t)262144)     // Wout bf16      131072
#define OFF_WSL    ((size_t)393216)     // Wslice bf16      4096
#define OFF_WM1    ((size_t)397312)     // Wm1 bf16       524288
#define OFF_WM2    ((size_t)921600)     // Wm2 bf16       524288
#define OFF_STOK   ((size_t)1445888)    // stok f32       262144 (atomic accum)
#define OFF_SNORM  ((size_t)1708032)    // snorm f32        8192 (atomic accum)
#define OFF_CHANS  ((size_t)1716224)    // chansum f32      4096 (atomic accum)
#define OFF_KV     ((size_t)1720320)    // kv f32          32768
#define OFF_SE     ((size_t)1753088)    // se f32           4096
#define OFF_OT     ((size_t)1757184)    // oT bf16        131072
#define OFF_FXT    ((size_t)1888256)    // fxT bf16  [bh][d][n]   67108864
#define OFF_SW     ((size_t)68997120)   // sw  bf16  [bh][n][g]  134217728
#define OFF_SWT    ((size_t)203214848)  // swT bf16  [bh][g][n]  134217728

// ==================== K0: weight conversion + accumulator zeroing ====================
__global__ void k0_prep(const float* __restrict__ Wx, const float* __restrict__ Wfx,
                        const float* __restrict__ Wout, const float* __restrict__ Wsl,
                        const float* __restrict__ Wm1, const float* __restrict__ Wm2,
                        unsigned short* __restrict__ wWx, unsigned short* __restrict__ wWfx,
                        unsigned short* __restrict__ wWout, unsigned short* __restrict__ wWsl,
                        unsigned short* __restrict__ wWm1, unsigned short* __restrict__ wWm2,
                        float* __restrict__ stok, float* __restrict__ snorm,
                        float* __restrict__ chans){
  int i = blockIdx.x*blockDim.x + threadIdx.x;
  int stride = gridDim.x*blockDim.x;
  for (int t=i; t<Cn*Cn; t+=stride){ wWx[t]=f2bf(Wx[t]); wWfx[t]=f2bf(Wfx[t]); wWout[t]=f2bf(Wout[t]); }
  for (int t=i; t<HIDn*Cn; t+=stride){ wWm1[t]=f2bf(Wm1[t]); wWm2[t]=f2bf(Wm2[t]); }
  for (int t=i; t<Gn*Dn; t+=stride) wWsl[t]=f2bf(Wsl[t]);
  for (int t=i; t<Bn*Hn*Gn*Dn; t+=stride) stok[t]=0.f;
  for (int t=i; t<Bn*Hn*Gn; t+=stride) snorm[t]=0.f;
  for (int t=i; t<Bn*Cn; t+=stride) chans[t]=0.f;
}

// ==================== K1: LN1 + xm/fx GEMMs + slice softmax (one wave / 16 tokens) ====================
__global__ __launch_bounds__(32) void k1_pass1(
    const float* __restrict__ x, const float* __restrict__ pos,
    const float* __restrict__ ln1g, const float* __restrict__ ln1b,
    const float* __restrict__ bx, const float* __restrict__ bfx,
    const float* __restrict__ bsl, const float* __restrict__ temp,
    const float* __restrict__ Wsb1, const float* __restrict__ bsb1,
    const float* __restrict__ Wsb2, const float* __restrict__ bsb2,
    const float* __restrict__ Wsb3, const float* __restrict__ bsb3,
    const unsigned short* __restrict__ wWx, const unsigned short* __restrict__ wWfx,
    const unsigned short* __restrict__ wWsl,
    unsigned short* __restrict__ swg, unsigned short* __restrict__ swT,
    unsigned short* __restrict__ fxT, float* __restrict__ snorm)
{
  __shared__ __align__(16) float          xbuf[16*Cn];   // 16KB
  __shared__ __align__(16) unsigned short xh  [16*Cn];   // 8KB  LN1(x) bf16
  __shared__ __align__(16) unsigned short xm  [16*Cn];   // 8KB
  __shared__ __align__(16) float          sbb [16*Gn];   // 4KB  spatial bias
  __shared__ __align__(16) float          tb0 [16*Gn];   // 4KB
  __shared__ __align__(16) float          tb1 [16*Gn];   // 4KB
  __shared__ __align__(16) float          lb  [16*Gn];   // 4KB  logits / sw f32
  __shared__ __align__(16) unsigned short swb [16*Gn];   // 2KB  sw bf16

  const int lane = threadIdx.x;
  const int half = lane >> 4;
  const int l15  = lane & 15;
  const int n0   = blockIdx.x * 16;       // global token index of tile
  const int b    = n0 / Nn;
  const int nloc = n0 - b*Nn;

  { // load x tile (coalesced float4)
    int row = lane>>1, c0 = (lane&1)*128;
    const float4* src = (const float4*)(x + (size_t)(n0+row)*Cn + c0);
    float4* dst = (float4*)(xbuf + row*Cn + c0);
    #pragma unroll
    for (int i=0;i<32;i++) dst[i]=src[i];
  }
  __syncthreads();
  if (lane < 16){ // LN1 -> bf16
    int row = lane;
    float m=0.f;
    for (int c=0;c<Cn;c++) m += xbuf[row*Cn+c];
    m *= (1.f/Cn);
    float v=0.f;
    for (int c=0;c<Cn;c++){ float d = xbuf[row*Cn+c]-m; v += d*d; }
    float r = rsqrtf(v*(1.f/Cn) + 1e-5f);
    for (int c=0;c<Cn;c++) xh[row*Cn+c] = f2bf((xbuf[row*Cn+c]-m)*r*ln1g[c] + ln1b[c]);
  }
  { // spatial-bias MLP: pos[4] -> 64 -> 64 -> G   (2 lanes per token)
    int row = lane>>1, seg = (lane&1)*32;
    float p0 = pos[(size_t)(n0+row)*4+0], p1 = pos[(size_t)(n0+row)*4+1];
    float p2 = pos[(size_t)(n0+row)*4+2], p3 = pos[(size_t)(n0+row)*4+3];
    for (int o=0;o<32;o++){
      int og = seg+o;
      float a = bsb1[og] + Wsb1[og*4+0]*p0 + Wsb1[og*4+1]*p1 + Wsb1[og*4+2]*p2 + Wsb1[og*4+3]*p3;
      tb0[row*64+og] = gelu_f(a);
    }
    __syncthreads();
    for (int o=0;o<32;o++){
      int og = seg+o; float a = bsb2[og];
      for (int k=0;k<64;k++) a += Wsb2[og*64+k]*tb0[row*64+k];
      tb1[row*64+og] = gelu_f(a);
    }
    __syncthreads();
    for (int o=0;o<32;o++){
      int og = seg+o; float a = bsb3[og];
      for (int k=0;k<64;k++) a += Wsb3[og*64+k]*tb1[row*64+k];
      sbb[row*64+og] = a;
    }
  }
  __syncthreads();

  // xm = LN1(x) @ Wx^T + bx  (bf16 to LDS)
  for (int j=0;j<16;j++){
    f32x8 c = fzero8();
    const unsigned short* arow = xh + l15*Cn;
    const unsigned short* brow = wWx + (size_t)(j*16 + l15)*Cn;
    for (int k0=0;k0<Cn;k0+=32) c = wmma_bf(load_a16(arow,k0,half), load_b16(brow,k0,half), c);
    int col = j*16 + l15; float bias = bx[col];
    #pragma unroll
    for (int i=0;i<8;i++){ int m = half*8+i; xm[m*Cn+col] = f2bf(c[i]+bias); }
  }
  // fx = LN1(x) @ Wfx^T + bfx  -> fxT[bh][d][n] bf16 (16B per-lane stores)
  for (int j=0;j<16;j++){
    f32x8 c = fzero8();
    const unsigned short* arow = xh + l15*Cn;
    const unsigned short* brow = wWfx + (size_t)(j*16 + l15)*Cn;
    for (int k0=0;k0<Cn;k0+=32) c = wmma_bf(load_a16(arow,k0,half), load_b16(brow,k0,half), c);
    int col = j*16 + l15; int h = col>>5, d = col&31;
    float bias = bfx[col];
    unsigned int w[4];
    #pragma unroll
    for (int i=0;i<4;i++) w[i] = pack2(f2bf(c[2*i]+bias), f2bf(c[2*i+1]+bias));
    uint4 u; u.x=w[0]; u.y=w[1]; u.z=w[2]; u.w=w[3];
    size_t off = ((size_t)(b*Hn+h)*Dn + d)*Nn + nloc + half*8;
    *(uint4*)(fxT+off) = u;
  }
  __syncthreads();

  // per-head slice logits + softmax + sw stores + snorm accumulation
  for (int h=0;h<Hn;h++){
    float tinv = 1.f / fmaxf(temp[h], 1e-4f);
    for (int gt=0; gt<4; gt++){
      f32x8 c = fzero8();
      const unsigned short* arow = xm + l15*Cn + h*Dn;
      const unsigned short* brow = wWsl + (size_t)(gt*16+l15)*Dn;
      c = wmma_bf(load_a16(arow,0,half), load_b16(brow,0,half), c);
      int g = gt*16+l15; float bias = bsl[g];
      #pragma unroll
      for (int i=0;i<8;i++){ int m=half*8+i; lb[m*64+g] = (c[i]+bias)*tinv + 0.1f*sbb[m*64+g]; }
    }
    __syncthreads();
    if (lane<16){ // softmax over G per token
      int row=lane;
      float mx=-1e30f;
      for (int g=0;g<64;g++) mx = fmaxf(mx, lb[row*64+g]);
      float s=0.f;
      for (int g=0;g<64;g++){ float e = __expf(lb[row*64+g]-mx); lb[row*64+g]=e; s+=e; }
      float inv = 1.f/s;
      for (int g=0;g<64;g++){ float w = lb[row*64+g]*inv; lb[row*64+g]=w; swb[row*64+g]=f2bf(w); }
    }
    __syncthreads();
    for (int g=lane; g<64; g+=32){ // snorm partial
      float s=0.f;
      for (int t=0;t<16;t++) s += lb[t*64+g];
      atomicAdd(snorm + (size_t)(b*Hn+h)*Gn + g, s);
    }
    { // sw [bh][n][g]
      int row = lane>>1, hg = (lane&1)*32;
      size_t off = ((size_t)(b*Hn+h)*Nn + nloc + row)*Gn + hg;
      const uint4* s4 = (const uint4*)(swb + row*64 + hg);
      uint4* d4 = (uint4*)(swg + off);
      #pragma unroll
      for (int q=0;q<4;q++) d4[q]=s4[q];
    }
    for (int g=lane; g<64; g+=32){ // swT [bh][g][n]
      unsigned int w[8];
      #pragma unroll
      for (int t=0;t<8;t++) w[t]=pack2(swb[(2*t)*64+g], swb[(2*t+1)*64+g]);
      size_t off = ((size_t)(b*Hn+h)*Gn + g)*Nn + nloc;
      uint4 u0; u0.x=w[0];u0.y=w[1];u0.z=w[2];u0.w=w[3];
      uint4 u1; u1.x=w[4];u1.y=w[5];u1.z=w[6];u1.w=w[7];
      *(uint4*)(swT+off)   = u0;
      *(uint4*)(swT+off+8) = u1;
    }
    __syncthreads();
  }
}

// ==================== K2: stok[bh] = sw^T @ fx, TDM tiles -> LDS, WMMA, atomic reduce ====================
#define K2CH 256   // K-chunk (tokens) per block

#if __has_builtin(__builtin_amdgcn_tensor_load_to_lds)
// Issue a 2-D TDM load: rows x K2CH bf16 tile, row stride Nn elements, into LDS at ldsOff.
// This toolchain exposes the 6-arg builtin: (u32x4 g0, i32x8 g1, i32x4 g2, i32x4 g3, i32x8, i32 cpol)
__device__ __forceinline__ void tdm_load_2d(const unsigned short* gsrc, unsigned int ldsOff, int rows){
  unsigned long long ga = (unsigned long long)(size_t)gsrc;
  // ---- D# group0: count=1 | lds_addr | global_addr(57b) | type=2 ----
  u32x4 g0 = { 1u,
               ldsOff,
               (unsigned int)(ga & 0xFFFFFFFFu),
               (unsigned int)((ga >> 32) & 0x1FFFFFFu) | (2u << 30) };
  // ---- D# group1 (256-bit LE bitfield) ----
  // [17:16]=data_size(1->2B); [79:48]=tensor_dim0=K2CH; [111:80]=tensor_dim1=rows;
  // [127:112]=tile_dim0=K2CH; [143:128]=tile_dim1=rows; [159:144]=tile_dim2=0;
  // [207:160]=tensor_dim0_stride=Nn; [255:208]=tensor_dim1_stride=0
  i32x8 g1 = { (int)(1u << 16),
               (int)((unsigned)(K2CH & 0xFFFF) << 16),
               (int)((unsigned)(rows & 0xFFFF) << 16),
               (int)((unsigned)(K2CH & 0xFFFF) << 16),
               (int)(rows & 0xFFFF),
               (int)Nn,
               0, 0 };
  i32x4 gz = {0,0,0,0};
  i32x8 gz8 = {0,0,0,0,0,0,0,0};
  __builtin_amdgcn_tensor_load_to_lds(g0, g1, gz, gz, gz8, 0);
}
#endif

__global__ __launch_bounds__(32) void k2_stok(
    const unsigned short* __restrict__ swT, const unsigned short* __restrict__ fxT,
    float* __restrict__ stok)
{
  __shared__ __align__(16) unsigned short As[Gn*K2CH];   // 32KB  sw^T tile [g][n]
  __shared__ __align__(16) unsigned short Bs[Dn*K2CH];   // 16KB  fx^T tile [d][n]
  const int lane = threadIdx.x, half=lane>>4, l15=lane&15;
  const int bh = blockIdx.y;
  const int nbase = blockIdx.x * K2CH;

#if __has_builtin(__builtin_amdgcn_tensor_load_to_lds)
  // Tensor Data Mover: stage both operand tiles into LDS asynchronously (TENSORcnt).
  tdm_load_2d(swT + (size_t)bh*Gn*Nn + nbase, (unsigned int)(size_t)&As[0], Gn);
  tdm_load_2d(fxT + (size_t)bh*Dn*Nn + nbase, (unsigned int)(size_t)&Bs[0], Dn);
  __builtin_amdgcn_s_wait_tensorcnt(0);
#else
  // Fallback: cooperative global->LDS copy (16B per lane)
  for (int idx=lane; idx<Gn*K2CH/8; idx+=32){
    int r = idx/(K2CH/8), cc = idx%(K2CH/8);
    ((uint4*)As)[idx] = *(const uint4*)(swT + ((size_t)bh*Gn + r)*Nn + nbase + cc*8);
  }
  for (int idx=lane; idx<Dn*K2CH/8; idx+=32){
    int r = idx/(K2CH/8), cc = idx%(K2CH/8);
    ((uint4*)Bs)[idx] = *(const uint4*)(fxT + ((size_t)bh*Dn + r)*Nn + nbase + cc*8);
  }
#endif
  __syncthreads();

  f32x8 acc[4][2];
  #pragma unroll
  for (int gt=0;gt<4;gt++)
    #pragma unroll
    for (int dt=0;dt<2;dt++) acc[gt][dt]=fzero8();

  for (int k0=0;k0<K2CH;k0+=32){
    v16bf A[4];
    #pragma unroll
    for (int gt=0;gt<4;gt++)
      A[gt] = load_a16(As + (gt*16 + l15)*K2CH, k0, half);
    v16bf Bm[2];
    #pragma unroll
    for (int dt=0;dt<2;dt++)
      Bm[dt] = load_b16(Bs + (dt*16 + l15)*K2CH, k0, half);
    #pragma unroll
    for (int gt=0;gt<4;gt++)
      #pragma unroll
      for (int dt=0;dt<2;dt++)
        acc[gt][dt] = wmma_bf(A[gt],Bm[dt],acc[gt][dt]);
  }
  #pragma unroll
  for (int gt=0;gt<4;gt++)
    #pragma unroll
    for (int dt=0;dt<2;dt++)
      #pragma unroll
      for (int i=0;i<8;i++)
        atomicAdd(stok + ((size_t)bh*Gn + gt*16 + half*8 + i)*Dn + dt*16 + l15, acc[gt][dt][i]);
}

// ==================== K3a: normalize stok by snorm, build kv = mean over heads ====================
__global__ __launch_bounds__(256) void k3a_norm(float* __restrict__ stok,
                                                const float* __restrict__ snorm,
                                                float* __restrict__ kv){
  int b=blockIdx.x, tid=threadIdx.x;
  for (int i=tid;i<Hn*Gn*Dn;i+=256){
    int h = i >> 11;           // / (G*D)
    int g = (i >> 5) & 63;
    float s = snorm[(size_t)(b*Hn+h)*Gn+g] + 1e-5f;
    stok[(size_t)b*Hn*Gn*Dn + i] /= s;
  }
  __syncthreads();
  for (int i=tid;i<Gn*Dn;i+=256){
    float s=0.f;
    for (int h=0;h<Hn;h++) s += stok[(size_t)(b*Hn+h)*Gn*Dn + i];
    kv[(size_t)b*Gn*Dn+i] = s*(1.f/Hn);
  }
}

// ==================== K3b: tiny attention over the G=64 slice tokens ====================
__global__ __launch_bounds__(128) void k3b_attn(
    const float* __restrict__ stok, const float* __restrict__ kv,
    const float* __restrict__ Wq, const float* __restrict__ Wk, const float* __restrict__ Wv,
    const float* __restrict__ ascale, const float* __restrict__ srs,
    unsigned short* __restrict__ oT)
{
  __shared__ float S[Gn*Dn], KVb[Gn*Dn], Q[Gn*Dn], Km[Gn*Dn], Vm[Gn*Dn], AW[Gn*Gn];
  int bh=blockIdx.x, b=bh/Hn, h=bh-b*Hn, tid=threadIdx.x;
  for (int i=tid;i<Gn*Dn;i+=128){ S[i]=stok[(size_t)bh*Gn*Dn+i]; KVb[i]=kv[(size_t)b*Gn*Dn+i]; }
  __syncthreads();
  for (int i=tid;i<Gn*Dn;i+=128){
    int g=i>>5, d=i&31;
    float q=0.f,kk=0.f,vv=0.f;
    for (int k2=0;k2<Dn;k2++){
      q  += S  [g*Dn+k2]*Wq[d*Dn+k2];
      kk += KVb[g*Dn+k2]*Wk[d*Dn+k2];
      vv += KVb[g*Dn+k2]*Wv[d*Dn+k2];
    }
    Q[i]=q; Km[i]=kk; Vm[i]=vv;
  }
  __syncthreads();
  if (tid < Gn){ // row L2 normalize q, k
    float s=0.f;
    for (int d=0;d<Dn;d++){ float q=Q[tid*Dn+d]; s+=q*q; }
    float inv = 1.f/fmaxf(sqrtf(s), 1e-12f);
    for (int d=0;d<Dn;d++) Q[tid*Dn+d]*=inv;
    s=0.f;
    for (int d=0;d<Dn;d++){ float k=Km[tid*Dn+d]; s+=k*k; }
    inv = 1.f/fmaxf(sqrtf(s), 1e-12f);
    for (int d=0;d<Dn;d++) Km[tid*Dn+d]*=inv;
  }
  __syncthreads();
  float sc = ascale[h];
  for (int i=tid;i<Gn*Gn;i+=128){
    int g=i>>6, g2=i&63;
    float s=0.f;
    for (int d=0;d<Dn;d++) s += Q[g*Dn+d]*Km[g2*Dn+d];
    AW[i]=s*sc;
  }
  __syncthreads();
  if (tid < Gn){
    float mx=-1e30f;
    for (int g2=0;g2<Gn;g2++) mx=fmaxf(mx, AW[tid*Gn+g2]);
    float s=0.f;
    for (int g2=0;g2<Gn;g2++){ float e=__expf(AW[tid*Gn+g2]-mx); AW[tid*Gn+g2]=e; s+=e; }
    float inv=1.f/s;
    for (int g2=0;g2<Gn;g2++) AW[tid*Gn+g2]*=inv;
  }
  __syncthreads();
  float sr = srs[0];
  for (int i=tid;i<Gn*Dn;i+=128){
    int g=i>>5, d=i&31;
    float s=0.f;
    for (int g2=0;g2<Gn;g2++) s += AW[g*Gn+g2]*Vm[g2*Dn+d];
    s += sr*S[i];
    oT[((size_t)bh*Dn + d)*Gn + g] = f2bf(s);   // store transposed [bh][d][g]
  }
}

// ==================== K4: ox + Wout + residual/LN + MLP + residual/LN + channel sums ====================
__global__ __launch_bounds__(32) void k4_pass2(
    const float* __restrict__ x,
    const unsigned short* __restrict__ swg, const unsigned short* __restrict__ oT,
    const unsigned short* __restrict__ wWout, const float* __restrict__ bout,
    const float* __restrict__ ln1pg, const float* __restrict__ ln1pb,
    const float* __restrict__ ln2g,  const float* __restrict__ ln2b,
    const unsigned short* __restrict__ wWm1, const float* __restrict__ bm1,
    const unsigned short* __restrict__ wWm2, const float* __restrict__ bm2,
    const float* __restrict__ ln2pg, const float* __restrict__ ln2pb,
    float* __restrict__ out, float* __restrict__ chans)
{
  __shared__ __align__(16) char sm[65536];     // exactly 64KB, phase-overlapped
  float*          abuf = (float*)sm;                        // [16][256] f32: a, then x2
  unsigned short* oxb  = (unsigned short*)(sm+16384);       // [16][256] bf16 (phase 1-2)
  unsigned short* xlb  = (unsigned short*)(sm+24576);       // [16][256] bf16 LN2(x2) (phase 3-4)
  unsigned short* hb   = (unsigned short*)(sm+32768);       // [16][1024] bf16 (phase 4-5)
  float*          mbuf = (float*)(sm+16384);                // [16][256] f32 (phase 5-6; reuses oxb+xlb)

  const int lane = threadIdx.x, half=lane>>4, l15=lane&15;
  const int n0 = blockIdx.x*16;
  const int b  = n0 / Nn;
  const int nloc = n0 - b*Nn;

  // step1: ox[n][h*D+d] = sum_g sw[n][g] * o[g][d]
  for (int h=0;h<Hn;h++){
    size_t bh = (size_t)b*Hn + h;
    for (int dt=0;dt<2;dt++){
      f32x8 c = fzero8();
      const unsigned short* arow = swg + (bh*Nn + nloc + l15)*Gn;
      const unsigned short* brow = oT + (bh*Dn + dt*16 + l15)*Gn;
      for (int k0=0;k0<Gn;k0+=32)
        c = wmma_bf(load_a16(arow,k0,half), load_b16(brow,k0,half), c);
      int col = h*Dn + dt*16 + l15;
      #pragma unroll
      for (int i=0;i<8;i++){ int m=half*8+i; oxb[m*Cn+col]=f2bf(c[i]); }
    }
  }
  __syncthreads();
  // step2: a = ox @ Wout^T + bout
  for (int j=0;j<16;j++){
    f32x8 c = fzero8();
    const unsigned short* arow = oxb + l15*Cn;
    const unsigned short* brow = wWout + (size_t)(j*16+l15)*Cn;
    for (int k0=0;k0<Cn;k0+=32)
      c = wmma_bf(load_a16(arow,k0,half), load_b16(brow,k0,half), c);
    int col=j*16+l15; float bias=bout[col];
    #pragma unroll
    for (int i=0;i<8;i++){ int m=half*8+i; abuf[m*Cn+col]=c[i]+bias; }
  }
  __syncthreads();
  // step3: x2 = x + LN1p(a); xlb = bf16(LN2(x2))
  if (lane<16){
    int row=lane;
    float m1=0.f;
    for (int c=0;c<Cn;c++) m1 += abuf[row*Cn+c];
    m1 *= (1.f/Cn);
    float v1=0.f;
    for (int c=0;c<Cn;c++){ float d=abuf[row*Cn+c]-m1; v1+=d*d; }
    float r1 = rsqrtf(v1*(1.f/Cn)+1e-5f);
    const float* xrow = x + (size_t)(n0+row)*Cn;
    for (int c=0;c<Cn;c++){
      float a=(abuf[row*Cn+c]-m1)*r1*ln1pg[c]+ln1pb[c];
      abuf[row*Cn+c]=xrow[c]+a;
    }
    float m2=0.f;
    for (int c=0;c<Cn;c++) m2 += abuf[row*Cn+c];
    m2 *= (1.f/Cn);
    float v2=0.f;
    for (int c=0;c<Cn;c++){ float d=abuf[row*Cn+c]-m2; v2+=d*d; }
    float r2 = rsqrtf(v2*(1.f/Cn)+1e-5f);
    for (int c=0;c<Cn;c++)
      xlb[row*Cn+c]=f2bf((abuf[row*Cn+c]-m2)*r2*ln2g[c]+ln2b[c]);
  }
  __syncthreads();
  // step4: h = gelu(LN2(x2) @ Wm1^T + bm1)
  for (int j=0;j<64;j++){
    f32x8 c = fzero8();
    const unsigned short* arow = xlb + l15*Cn;
    const unsigned short* brow = wWm1 + (size_t)(j*16+l15)*Cn;
    for (int k0=0;k0<Cn;k0+=32){
      __builtin_prefetch(brow + k0 + 64, 0, 1);
      c = wmma_bf(load_a16(arow,k0,half), load_b16(brow,k0,half), c);
    }
    int col=j*16+l15; float bias=bm1[col];
    #pragma unroll
    for (int i=0;i<8;i++){ int m=half*8+i; hb[m*HIDn+col]=f2bf(gelu_f(c[i]+bias)); }
  }
  __syncthreads();
  // step5: m = h @ Wm2^T + bm2
  for (int j=0;j<16;j++){
    f32x8 c = fzero8();
    const unsigned short* arow = hb + l15*HIDn;
    const unsigned short* brow = wWm2 + (size_t)(j*16+l15)*HIDn;
    for (int k0=0;k0<HIDn;k0+=32){
      __builtin_prefetch(brow + k0 + 64, 0, 1);
      c = wmma_bf(load_a16(arow,k0,half), load_b16(brow,k0,half), c);
    }
    int col=j*16+l15; float bias=bm2[col];
    #pragma unroll
    for (int i=0;i<8;i++){ int m=half*8+i; mbuf[m*Cn+col]=c[i]+bias; }
  }
  __syncthreads();
  // step6: x3 = x2 + LN2p(m) -> global out + mbuf (for SE channel sums)
  if (lane<16){
    int row=lane;
    float mm=0.f;
    for (int c=0;c<Cn;c++) mm += mbuf[row*Cn+c];
    mm *= (1.f/Cn);
    float vv=0.f;
    for (int c=0;c<Cn;c++){ float d=mbuf[row*Cn+c]-mm; vv+=d*d; }
    float rr = rsqrtf(vv*(1.f/Cn)+1e-5f);
    float* orow = out + (size_t)(n0+row)*Cn;
    for (int c=0;c<Cn;c++){
      float mn=(mbuf[row*Cn+c]-mm)*rr*ln2pg[c]+ln2pb[c];
      float x3=abuf[row*Cn+c]+mn;
      mbuf[row*Cn+c]=x3;
      orow[c]=x3;
    }
  }
  __syncthreads();
  for (int c=lane;c<Cn;c+=32){  // per-block channel partial sums
    float s=0.f;
    for (int t=0;t<16;t++) s += mbuf[t*Cn+c];
    atomicAdd(chans + b*Cn + c, s);
  }
}

// ==================== K5: SE gate from mean-pooled x3 ====================
__global__ __launch_bounds__(64) void k5_se(const float* __restrict__ chans,
                                            const float* __restrict__ Wse1, const float* __restrict__ bse1,
                                            const float* __restrict__ Wse2, const float* __restrict__ bse2,
                                            float* __restrict__ se){
  __shared__ float t1[64];
  int b=blockIdx.x, tid=threadIdx.x;
  float a=bse1[tid];
  for (int c=0;c<Cn;c++) a += (chans[b*Cn+c]*(1.f/Nn))*Wse1[tid*Cn+c];
  t1[tid]=gelu_f(a);
  __syncthreads();
  for (int c=tid;c<Cn;c+=64){
    float s=bse2[c];
    for (int k=0;k<64;k++) s += t1[k]*Wse2[c*64+k];
    se[b*Cn+c]=s;
  }
}

// ==================== K6: out = x3 * (1 + se) (in place) ====================
__global__ void k6_scale(float* __restrict__ out, const float* __restrict__ se){
  size_t i = (size_t)blockIdx.x*blockDim.x + threadIdx.x;
  size_t stride = (size_t)gridDim.x*blockDim.x;
  const size_t total = (size_t)BNn*Cn;
  for (size_t t=i; t<total; t+=stride){
    int c = (int)(t & (Cn-1));
    int b = (int)(t / ((size_t)Nn*Cn));
    out[t] = out[t]*(1.f + se[b*Cn+c]);
  }
}

// ==================== host-side launch ====================
extern "C" void kernel_launch(void* const* d_in, const int* in_sizes, int n_in,
                              void* d_out, int out_size, void* d_ws, size_t ws_size,
                              hipStream_t stream){
  (void)in_sizes; (void)n_in; (void)out_size; (void)ws_size;
  const float* x    =(const float*)d_in[0];
  const float* pos  =(const float*)d_in[1];
  const float* ln1g =(const float*)d_in[2];  const float* ln1b =(const float*)d_in[3];
  const float* ln1pg=(const float*)d_in[4];  const float* ln1pb=(const float*)d_in[5];
  const float* ln2g =(const float*)d_in[6];  const float* ln2b =(const float*)d_in[7];
  const float* ln2pg=(const float*)d_in[8];  const float* ln2pb=(const float*)d_in[9];
  const float* Wx   =(const float*)d_in[10]; const float* bx   =(const float*)d_in[11];
  const float* Wfx  =(const float*)d_in[12]; const float* bfx  =(const float*)d_in[13];
  const float* Wsl  =(const float*)d_in[14]; const float* bsl  =(const float*)d_in[15];
  const float* temp =(const float*)d_in[16];
  const float* Wq   =(const float*)d_in[17]; const float* Wk   =(const float*)d_in[18];
  const float* Wv   =(const float*)d_in[19];
  const float* ascl =(const float*)d_in[20]; const float* srs  =(const float*)d_in[21];
  const float* Wout =(const float*)d_in[22]; const float* bout =(const float*)d_in[23];
  const float* Wm1  =(const float*)d_in[24]; const float* bm1  =(const float*)d_in[25];
  const float* Wm2  =(const float*)d_in[26]; const float* bm2  =(const float*)d_in[27];
  const float* Wsb1 =(const float*)d_in[28]; const float* bsb1 =(const float*)d_in[29];
  const float* Wsb2 =(const float*)d_in[30]; const float* bsb2 =(const float*)d_in[31];
  const float* Wsb3 =(const float*)d_in[32]; const float* bsb3 =(const float*)d_in[33];
  const float* Wse1 =(const float*)d_in[34]; const float* bse1 =(const float*)d_in[35];
  const float* Wse2 =(const float*)d_in[36]; const float* bse2 =(const float*)d_in[37];
  float* out = (float*)d_out;
  char*  ws  = (char*)d_ws;

  unsigned short* wWx  =(unsigned short*)(ws+OFF_WX);
  unsigned short* wWfx =(unsigned short*)(ws+OFF_WFX);
  unsigned short* wWout=(unsigned short*)(ws+OFF_WOUT);
  unsigned short* wWsl =(unsigned short*)(ws+OFF_WSL);
  unsigned short* wWm1 =(unsigned short*)(ws+OFF_WM1);
  unsigned short* wWm2 =(unsigned short*)(ws+OFF_WM2);
  float* stok =(float*)(ws+OFF_STOK);
  float* snorm=(float*)(ws+OFF_SNORM);
  float* chans=(float*)(ws+OFF_CHANS);
  float* kv   =(float*)(ws+OFF_KV);
  float* se   =(float*)(ws+OFF_SE);
  unsigned short* oT  =(unsigned short*)(ws+OFF_OT);
  unsigned short* fxT =(unsigned short*)(ws+OFF_FXT);
  unsigned short* swg =(unsigned short*)(ws+OFF_SW);
  unsigned short* swT =(unsigned short*)(ws+OFF_SWT);

  k0_prep<<<512,256,0,stream>>>(Wx,Wfx,Wout,Wsl,Wm1,Wm2,
                                wWx,wWfx,wWout,wWsl,wWm1,wWm2,
                                stok,snorm,chans);
  k1_pass1<<<BNn/16,32,0,stream>>>(x,pos,ln1g,ln1b,bx,bfx,bsl,temp,
                                   Wsb1,bsb1,Wsb2,bsb2,Wsb3,bsb3,
                                   wWx,wWfx,wWsl,swg,swT,fxT,snorm);
  k2_stok<<<dim3(Nn/K2CH, Bn*Hn),32,0,stream>>>(swT,fxT,stok);
  k3a_norm<<<Bn,256,0,stream>>>(stok,snorm,kv);
  k3b_attn<<<Bn*Hn,128,0,stream>>>(stok,kv,Wq,Wk,Wv,ascl,srs,oT);
  k4_pass2<<<BNn/16,32,0,stream>>>(x,swg,oT,wWout,bout,ln1pg,ln1pb,ln2g,ln2b,
                                   wWm1,bm1,wWm2,bm2,ln2pg,ln2pb,out,chans);
  k5_se<<<Bn,64,0,stream>>>(chans,Wse1,bse1,Wse2,bse2,se);
  k6_scale<<<4096,256,0,stream>>>(out,se);
}
